// MultivariateTSModel_54528904790728
// MI455X (gfx1250) — matrix operand order repaired and checked by reading, not compiled, not run on hardware
//
#include <hip/hip_runtime.h>
#include <hip/hip_bf16.h>
#include <math.h>

typedef __attribute__((ext_vector_type(16))) _Float16 v16h;
typedef __attribute__((ext_vector_type(8)))  float    v8f;

#define DEVFN static __device__ __forceinline__

DEVFN float sigm(float v) { return 1.0f / (1.0f + __expf(-v)); }

DEVFN v8f wmma_f16(v16h a, v16h b, v8f c) {
    // D = A(16x32 f16) * B(32x16 f16) + C(16x16 f32)
    return __builtin_amdgcn_wmma_f32_16x16x32_f16(
        /*neg_a=*/false, a, /*neg_b=*/false, b,
        /*c_mod=*/(short)0, c, /*reuse_a=*/false, /*reuse_b=*/false);
}

// ---------------- CDNA5 async global->LDS copy (ASYNCcnt path) ---------------
typedef int v4i_ __attribute__((vector_size(16)));
typedef __attribute__((address_space(1))) v4i_* g4_t;   // global int4*
typedef __attribute__((address_space(3))) v4i_* l4_t;   // LDS int4*

DEVFN void async_b128(const void* g, void* l) {
#if __has_builtin(__builtin_amdgcn_global_load_async_to_lds_b128)
    __builtin_amdgcn_global_load_async_to_lds_b128((g4_t)g, (l4_t)l, 0, 0);
#else
    unsigned ldsoff = (unsigned)(unsigned long long)(l4_t)l;
    unsigned long long ga = (unsigned long long)g;
    asm volatile("global_load_async_to_lds_b128 %0, %1, off"
                 :: "v"(ldsoff), "v"(ga) : "memory");
#endif
}

template <int N>
DEVFN void wait_async() {
#if __has_builtin(__builtin_amdgcn_s_wait_asynccnt)
    __builtin_amdgcn_s_wait_asynccnt((short)N);
#else
    asm volatile("s_wait_asynccnt %0" :: "i"(N) : "memory");
#endif
}

// ---------------- WMMA fragment loaders (contiguous k order: element s holds k = 16*(lane/16) + s) -----------------
// A fragment: 16x32 (MxK) f16, row-major source, leading dim ld.
DEVFN v16h load_A16(const _Float16* base, int ld) {
    int lane = threadIdx.x & 31;
    int m  = lane & 15;
    int kh = (lane >> 4) & 1;
    v16h a;
#pragma unroll
    for (int i = 0; i < 8; ++i) {
        int k = kh * 16 + 2 * i;
        a[2 * i]     = base[m * ld + k];
        a[2 * i + 1] = base[m * ld + k + 1];
    }
    return a;
}

// Same, but source is f32 in LDS (convert at fragment load).
DEVFN v16h load_A16_f32(const float* base, int ld) {
    int lane = threadIdx.x & 31;
    int m  = lane & 15;
    int kh = (lane >> 4) & 1;
    v16h a;
#pragma unroll
    for (int i = 0; i < 8; ++i) {
        int k = kh * 16 + 2 * i;
        a[2 * i]     = (_Float16)base[m * ld + k];
        a[2 * i + 1] = (_Float16)base[m * ld + k + 1];
    }
    return a;
}

// B fragment for B[k][n] = W[n][k] (X @ W^T), W row-major: 16 contiguous halves/lane.
DEVFN v16h load_B16_wrow(const _Float16* Wbase, int ldw) {
    int lane = threadIdx.x & 31;
    int n  = lane & 15;
    int kb = (lane >> 4) << 4;
    v16h b;
#pragma unroll
    for (int i = 0; i < 16; ++i) b[i] = Wbase[n * ldw + kb + i];
    return b;
}

// B fragment for B stored [K][N] row-major (direct).
DEVFN v16h load_B16_kn(const _Float16* Bbase, int ldb) {
    int lane = threadIdx.x & 31;
    int n  = lane & 15;
    int kb = (lane >> 4) << 4;
    v16h b;
#pragma unroll
    for (int i = 0; i < 16; ++i) b[i] = Bbase[(kb + i) * ldb + n];
    return b;
}

// ---------------------------------------------------------------------------
// Kernel 1: embedding + GRU + attention scan. One block per batch, 8 waves.
// All state + f16 weights live in LDS (~200KB, fits CDNA5's 320KB/WGP).
// ---------------------------------------------------------------------------
static constexpr int TT = 256, CC = 64, EE = 32, HH = 64, G3 = 192;

static constexpr size_t SCAN_HALFS =
    (size_t)G3 * EE + (size_t)G3 * HH + (size_t)CC * HH + (size_t)CC * HH +
    (size_t)CC * HH + (size_t)CC * EE + (size_t)CC * HH + (size_t)CC * HH +
    (size_t)CC * HH;
static constexpr size_t SCAN_FLOATS =
    (size_t)CC * HH + (size_t)CC * G3 + (size_t)CC * HH + (size_t)CC * CC +
    (size_t)CC * CC + G3 + G3 + CC + CC + CC + 4;
static constexpr size_t SCAN_SMEM = SCAN_HALFS * 2 + SCAN_FLOATS * 4;

__global__ __launch_bounds__(256)
void k_scan(const float* __restrict__ x,
            const float* __restrict__ embed_w, const float* __restrict__ embed_b,
            const float* __restrict__ W_ih, const float* __restrict__ b_ih,
            const float* __restrict__ W_hh, const float* __restrict__ b_hh,
            const float* __restrict__ Wq, const float* __restrict__ bq,
            const float* __restrict__ Wk, const float* __restrict__ bk,
            const float* __restrict__ gate_bias,
            const float* __restrict__ Wp, const float* __restrict__ bp,
            float* __restrict__ out_mix, float* __restrict__ out_dnc,
            float* __restrict__ out_pred)
{
    extern __shared__ char smem[];
    _Float16* sWih = (_Float16*)smem;           // [192][32]
    _Float16* sWhh = sWih + G3 * EE;            // [192][64]
    _Float16* sWq  = sWhh + G3 * HH;            // [64][64]
    _Float16* sWk  = sWq  + CC * HH;            // [64][64]
    _Float16* sH16 = sWk  + CC * HH;            // [64][64]
    _Float16* sEmb = sH16 + CC * HH;            // [64][32]
    _Float16* sQ16 = sEmb + CC * EE;            // [64][64]
    _Float16* sK16 = sQ16 + CC * HH;            // [64][64]
    _Float16* sS16 = sK16 + CC * HH;            // [64][64]
    float* sH   = (float*)(sS16 + CC * HH);     // [64][64]
    float* sG   = sH   + CC * HH;               // [64][192]
    float* sGn  = sG   + CC * G3;               // [64][64]
    float* sS   = sGn  + CC * HH;               // [64][64]
    float* sGB  = sS   + CC * CC;               // [64][64]
    float* sBih = sGB  + CC * CC;               // [192]
    float* sBhh = sBih + G3;                    // [192]
    float* sBq  = sBhh + G3;                    // [64]
    float* sBk  = sBq  + CC;                    // [64]
    float* sWp_ = sBk  + CC;                    // [64]
    float* sNorm = sWp_ + CC;                   // [1]

    const int tid  = threadIdx.x;
    const int b    = blockIdx.x;
    const int wave = tid >> 5;
    const int lane = tid & 31;

    for (int i = tid; i < G3 * EE; i += 256) sWih[i] = (_Float16)W_ih[i];
    for (int i = tid; i < G3 * HH; i += 256) sWhh[i] = (_Float16)W_hh[i];
    for (int i = tid; i < CC * HH; i += 256) {
        sWq[i] = (_Float16)Wq[i];
        sWk[i] = (_Float16)Wk[i];
        sGB[i] = gate_bias[i];
        sH[i]  = 0.0f;
        sH16[i] = (_Float16)0.0f;
    }
    for (int i = tid; i < G3; i += 256) { sBih[i] = b_ih[i]; sBhh[i] = b_hh[i]; }
    if (tid < CC) { sBq[tid] = bq[tid]; sBk[tid] = bk[tid]; sWp_[tid] = Wp[tid]; }
    const float bpv = bp[0];

    float dnc[16];
#pragma unroll
    for (int p = 0; p < 16; ++p) dnc[p] = 0.0f;
    __syncthreads();

    for (int t = 0; t < TT; ++t) {
        // embedding
        for (int i = tid; i < CC * EE; i += 256) {
            int c = i >> 5, e = i & 31;
            float xv = x[((size_t)b * TT + t) * CC + c];
            sEmb[i] = (_Float16)(xv * embed_w[e] + embed_b[e]);
        }
        if (tid == 0) sNorm[0] = 0.0f;
        __syncthreads();

        // gates: gi = emb@W_ih^T, gh = h@W_hh^T (n-gate kept split)
#pragma unroll
        for (int it = 0; it < 6; ++it) {
            int jt = wave + (it << 3);
            int mi = jt & 3, nj = jt >> 2;
            v16h a0 = load_A16(sEmb + mi * 16 * EE, EE);
            v8f ai = {};
            ai = wmma_f16(a0, load_B16_wrow(sWih + (nj * 16) * EE, EE), ai);
            v8f ah = {};
#pragma unroll
            for (int kk = 0; kk < HH; kk += 32) {
                ah = wmma_f16(load_A16(sH16 + mi * 16 * HH + kk, HH),
                              load_B16_wrow(sWhh + (nj * 16) * HH + kk, HH), ah);
            }
            int colg  = nj * 16 + (lane & 15);
            int rbase = ((lane >> 4) << 3) + mi * 16;
            if (nj < 8) {
#pragma unroll
                for (int r = 0; r < 8; ++r)
                    sG[(rbase + r) * G3 + colg] = ai[r] + ah[r] + sBih[colg] + sBhh[colg];
            } else {
#pragma unroll
                for (int r = 0; r < 8; ++r) {
                    sG[(rbase + r) * G3 + colg]          = ai[r] + sBih[colg];
                    sGn[(rbase + r) * HH + (colg - 128)] = ah[r] + sBhh[colg];
                }
            }
        }
        __syncthreads();

        // GRU elementwise -> hg
#pragma unroll
        for (int p = 0; p < 16; ++p) {
            int idx = p * 256 + tid;
            int c = idx >> 6, hh = idx & 63;
            float r  = sigm(sG[c * G3 + hh]);
            float z  = sigm(sG[c * G3 + 64 + hh]);
            float nn = tanhf(sG[c * G3 + 128 + hh] + r * sGn[idx]);
            float hg = (1.0f - z) * nn + z * sH[idx];
            sH[idx]   = hg;
            sH16[idx] = (_Float16)hg;
        }
        __syncthreads();

        // q / k projections
#pragma unroll
        for (int it = 0; it < 4; ++it) {
            int jt = wave + (it << 3);
            int which = jt >> 4;
            int rem = jt & 15;
            int mi = rem & 3, nj = rem >> 2;
            const _Float16* Wm = which ? sWk : sWq;
            v8f acc = {};
#pragma unroll
            for (int kk = 0; kk < HH; kk += 32) {
                acc = wmma_f16(load_A16(sH16 + mi * 16 * HH + kk, HH),
                               load_B16_wrow(Wm + (nj * 16) * HH + kk, HH), acc);
            }
            _Float16* dst = which ? sK16 : sQ16;
            const float* bias = which ? sBk : sBq;
            int col = nj * 16 + (lane & 15);
            int rbase = ((lane >> 4) << 3) + mi * 16;
#pragma unroll
            for (int r = 0; r < 8; ++r)
                dst[(rbase + r) * HH + col] = (_Float16)(acc[r] + bias[col]);
        }
        __syncthreads();

        // scores = q @ k^T + Frobenius reduction
        float ssq = 0.0f;
#pragma unroll
        for (int it = 0; it < 2; ++it) {
            int jt = wave + (it << 3);
            int mi = jt & 3, nj = jt >> 2;
            v8f acc = {};
#pragma unroll
            for (int kk = 0; kk < 64; kk += 32) {
                acc = wmma_f16(load_A16(sQ16 + mi * 16 * 64 + kk, 64),
                               load_B16_wrow(sK16 + (nj * 16) * 64 + kk, 64), acc);
            }
            int col = nj * 16 + (lane & 15);
            int rbase = ((lane >> 4) << 3) + mi * 16;
#pragma unroll
            for (int r = 0; r < 8; ++r) {
                sS[(rbase + r) * 64 + col] = acc[r];
                ssq += acc[r] * acc[r];
            }
        }
        atomicAdd(sNorm, ssq);   // ds_add_f32
        __syncthreads();

        // transfer = tanh(s/||s||_F) * sigmoid(|.|+gate_bias); emit mix
        float inv = rsqrtf(sNorm[0] + 1e-30f);
        float* mixp = out_mix + ((size_t)b * TT + t) * (CC * CC);
#pragma unroll
        for (int p = 0; p < 16; ++p) {
            int idx = p * 256 + tid;
            float s  = tanhf(sS[idx] * inv);
            float g  = sigm(fabsf(s) + sGB[idx]);
            float tr = s * g;
            sS16[idx] = (_Float16)tr;
            mixp[idx] = tr;
            dnc[p] += tr;
        }
        __syncthreads();

        // h_new = transfer @ hg
        v8f hn0 = {}, hn1 = {};
        const int jt0 = wave, jt1 = wave + 8;
        {
            int mi = jt0 & 3, nj = jt0 >> 2;
#pragma unroll
            for (int kk = 0; kk < 64; kk += 32)
                hn0 = wmma_f16(load_A16(sS16 + mi * 16 * 64 + kk, 64),
                               load_B16_kn(sH16 + kk * 64 + nj * 16, 64), hn0);
        }
        {
            int mi = jt1 & 3, nj = jt1 >> 2;
#pragma unroll
            for (int kk = 0; kk < 64; kk += 32)
                hn1 = wmma_f16(load_A16(sS16 + mi * 16 * 64 + kk, 64),
                               load_B16_kn(sH16 + kk * 64 + nj * 16, 64), hn1);
        }
        __syncthreads();
        {
            int mi = jt0 & 3, nj = jt0 >> 2;
            int col = nj * 16 + (lane & 15);
            int rbase = ((lane >> 4) << 3) + mi * 16;
#pragma unroll
            for (int r = 0; r < 8; ++r) {
                sH[(rbase + r) * 64 + col]   = hn0[r];
                sH16[(rbase + r) * 64 + col] = (_Float16)hn0[r];
            }
            mi = jt1 & 3; nj = jt1 >> 2;
            col = nj * 16 + (lane & 15);
            rbase = ((lane >> 4) << 3) + mi * 16;
#pragma unroll
            for (int r = 0; r < 8; ++r) {
                sH[(rbase + r) * 64 + col]   = hn1[r];
                sH16[(rbase + r) * 64 + col] = (_Float16)hn1[r];
            }
        }
        __syncthreads();

        // predicted[b, t-1, c] = h_new[c,:]·Wp + bp (hs[:,1:])
        if (t >= 1 && tid < CC) {
            float acc = bpv;
#pragma unroll
            for (int hh = 0; hh < 64; ++hh) acc += sH[tid * 64 + hh] * sWp_[hh];
            out_pred[((size_t)b * (TT - 1) + (t - 1)) * CC + tid] = acc;
        }
        __syncthreads();
    }

#pragma unroll
    for (int p = 0; p < 16; ++p) {
        int idx = p * 256 + tid;
        out_dnc[(size_t)b * 4096 + idx] = dnc[p] * (1.0f / 256.0f);
    }
}

// ---------------------------------------------------------------------------
// Kernel 2: async-staged, double-buffered WMMA GEMM.
//   Out = relu(A @ W^T + bias) -> f16.  Block tile 128x128, BK=64.
//   8 waves, each 32(M)x64(N): 2 A-frags x 4 B-frags -> 16 WMMA per K-tile.
//   Tiles staged with GLOBAL_LOAD_ASYNC_TO_LDS_B128 (ASYNCcnt, in-order),
//   so s_wait_asynccnt <N_next_issued> releases the current buffer.
// ---------------------------------------------------------------------------
template <int AF16>
__global__ __launch_bounds__(256)
void k_gemm_async(const void* __restrict__ Av, const _Float16* __restrict__ Wh,
                  const float* __restrict__ bias, _Float16* __restrict__ Out,
                  int M, int N, int K)
{
    constexpr int BM = 128, BN = 128, BK = 64;
    constexpr int ABYTES = AF16 ? (BM * BK * 2) : (BM * BK * 4);
    constexpr int BBYTES = BN * BK * 2;
    constexpr int NASYNC = (ABYTES + BBYTES) / (256 * 16); // async instrs/wave/tile
    constexpr int ACHUNKS = ABYTES / 16;                   // 16B chunks in A tile
    constexpr int APT = ACHUNKS / 256;                     // chunks per thread
    constexpr int ACPR = (AF16 ? BK * 2 : BK * 4) / 16;    // 16B chunks per A row

    extern __shared__ char gsm[];
    char*     sA0 = gsm;
    char*     sA1 = gsm + ABYTES;
    _Float16* sB0 = (_Float16*)(gsm + 2 * (size_t)ABYTES);
    _Float16* sB1 = sB0 + BN * BK;

    const int tid  = threadIdx.x;
    const int bm   = blockIdx.x * BM;
    const int bn   = blockIdx.y * BN;
    const int wave = tid >> 5, lane = tid & 31;
    const int wm   = (wave & 3) * 32;   // wave M offset in block tile
    const int wn   = (wave >> 2) * 64;  // wave N offset in block tile

    const float*    Af = (const float*)Av;
    const _Float16* Ah = (const _Float16*)Av;

    auto issue_tile = [&](int kk, char* dA, _Float16* dB) {
#pragma unroll
        for (int i = 0; i < APT; ++i) {
            int idx = tid + i * 256;
            int r = idx / ACPR, c = idx % ACPR;
            const void* g = AF16
                ? (const void*)(Ah + (size_t)(bm + r) * K + kk + c * 8)
                : (const void*)(Af + (size_t)(bm + r) * K + kk + c * 4);
            async_b128(g, dA + (size_t)r * (ACPR * 16) + c * 16);
        }
#pragma unroll
        for (int i = 0; i < 4; ++i) {
            int idx = tid + i * 256;
            int r = idx >> 3, c = idx & 7;  // 8 x 16B chunks per 64-half row
            async_b128(Wh + (size_t)(bn + r) * K + kk + c * 8,
                       (char*)(dB + (size_t)r * BK + c * 8));
        }
    };

    v8f zero = {};
    v8f acc[2][4];
#pragma unroll
    for (int i = 0; i < 2; ++i)
#pragma unroll
        for (int j = 0; j < 4; ++j) acc[i][j] = zero;

    issue_tile(0, sA0, sB0);
    const int nt = K / BK;
    for (int it = 0; it < nt; ++it) {
        char*     curA = (it & 1) ? sA1 : sA0;
        _Float16* curB = (it & 1) ? sB1 : sB0;
        if (it + 1 < nt) {
            issue_tile((it + 1) * BK, (it & 1) ? sA0 : sA1, (it & 1) ? sB0 : sB1);
            wait_async<NASYNC>();   // current buffer's loads complete (in-order)
        } else {
            wait_async<0>();
        }
        __syncthreads();

#pragma unroll
        for (int ks = 0; ks < BK; ks += 32) {
            v16h a0, a1;
            if (AF16) {
                const _Float16* At = (const _Float16*)curA;
                a0 = load_A16(At + (wm + 0)  * BK + ks, BK);
                a1 = load_A16(At + (wm + 16) * BK + ks, BK);
            } else {
                const float* At = (const float*)curA;
                a0 = load_A16_f32(At + (wm + 0)  * BK + ks, BK);
                a1 = load_A16_f32(At + (wm + 16) * BK + ks, BK);
            }
#pragma unroll
            for (int bi = 0; bi < 4; ++bi) {
                v16h bf = load_B16_wrow(curB + (wn + bi * 16) * BK + ks, BK);
                acc[0][bi] = wmma_f16(a0, bf, acc[0][bi]);
                acc[1][bi] = wmma_f16(a1, bf, acc[1][bi]);
            }
        }
        __syncthreads();
    }

    const int col = lane & 15;
    const int rb  = (lane >> 4) << 3;
#pragma unroll
    for (int ai = 0; ai < 2; ++ai) {
#pragma unroll
        for (int bi = 0; bi < 4; ++bi) {
#pragma unroll
            for (int r = 0; r < 8; ++r) {
                int gr = bm + wm + ai * 16 + rb + r;
                int gc = bn + wn + bi * 16 + col;
                float v = acc[ai][bi][r] + bias[gc];
                v = v > 0.0f ? v : 0.0f;
                Out[(size_t)gr * N + gc] = (_Float16)v;
            }
        }
    }
}

__global__ void k_f32_to_f16(const float* __restrict__ src, _Float16* __restrict__ dst, int n)
{
    int i = blockIdx.x * blockDim.x + threadIdx.x;
    if (i < n) dst[i] = (_Float16)src[i];
}

// logits[b] = b3 + mean_t (h2[b,t] @ W3^T)
__global__ __launch_bounds__(256)
void k_logits(const _Float16* __restrict__ h2, const float* __restrict__ W3,
              const float* __restrict__ b3, float* __restrict__ logits)
{
    __shared__ float red0[256];
    __shared__ float red1[256];
    const int b = blockIdx.x, tid = threadIdx.x;
    float a0 = 0.0f, a1 = 0.0f;
    for (int t = 0; t < 256; ++t) {
        const _Float16* row = h2 + ((size_t)b * 256 + t) * 1024;
        for (int k = tid; k < 1024; k += 256) {
            float hv = (float)row[k];
            a0 += hv * W3[k];
            a1 += hv * W3[1024 + k];
        }
    }
    red0[tid] = a0; red1[tid] = a1;
    __syncthreads();
    for (int s = 128; s > 0; s >>= 1) {
        if (tid < s) { red0[tid] += red0[tid + s]; red1[tid] += red1[tid + s]; }
        __syncthreads();
    }
    if (tid == 0) {
        logits[b * 2 + 0] = b3[0] + red0[0] * (1.0f / 256.0f);
        logits[b * 2 + 1] = b3[1] + red1[0] * (1.0f / 256.0f);
    }
}

__global__ void k_xslice(const float* __restrict__ x, float* __restrict__ out)
{
    int i = blockIdx.x * blockDim.x + threadIdx.x;
    if (i >= 32 * 255 * 64) return;
    int c = i & 63;
    int rem = i >> 6;
    int t = rem % 255;
    int b = rem / 255;
    out[i] = x[((size_t)b * 256 + (t + 1)) * 64 + c];
}

extern "C" void kernel_launch(void* const* d_in, const int* in_sizes, int n_in,
                              void* d_out, int out_size, void* d_ws, size_t ws_size,
                              hipStream_t stream)
{
    const float* x     = (const float*)d_in[0];
    const float* ew    = (const float*)d_in[1];
    const float* eb    = (const float*)d_in[2];
    const float* W_ih  = (const float*)d_in[3];
    const float* b_ih  = (const float*)d_in[4];
    const float* W_hh  = (const float*)d_in[5];
    const float* b_hh  = (const float*)d_in[6];
    const float* Wq    = (const float*)d_in[7];
    const float* bq    = (const float*)d_in[8];
    const float* Wk    = (const float*)d_in[9];
    const float* bk    = (const float*)d_in[10];
    const float* gateb = (const float*)d_in[11];
    const float* Wp    = (const float*)d_in[12];
    const float* bp    = (const float*)d_in[13];
    const float* W1    = (const float*)d_in[14];
    const float* b1    = (const float*)d_in[15];
    const float* W2    = (const float*)d_in[16];
    const float* b2    = (const float*)d_in[17];
    const float* W3    = (const float*)d_in[18];
    const float* b3    = (const float*)d_in[19];

    // Output tuple: logits[64] | DNC[131072] | mix[33554432] | predicted[522240] | x[:,1:,:][522240]
    float* out      = (float*)d_out;
    float* o_logits = out;
    float* o_dnc    = out + 64;
    float* o_mix    = o_dnc + (size_t)32 * 64 * 64;
    float* o_pred   = o_mix + (size_t)32 * 256 * 64 * 64;
    float* o_xs     = o_pred + (size_t)32 * 255 * 64;

    // Workspace: f16 W1 | f16 W2 | f16 h1[8192,2048] | f16 h2[8192,1024]
    char* ws = (char*)d_ws;
    _Float16* W1h = (_Float16*)ws;
    _Float16* W2h = (_Float16*)(ws + (size_t)16777216);
    _Float16* h1  = (_Float16*)(ws + (size_t)16777216 + 4194304);
    _Float16* h2  = (_Float16*)(ws + (size_t)16777216 + 4194304 + 33554432);

    // 1) scan (writes mix, DNC, predicted)
    k_scan<<<dim3(32), dim3(256), SCAN_SMEM, stream>>>(
        x, ew, eb, W_ih, b_ih, W_hh, b_hh, Wq, bq, Wk, bk, gateb, Wp, bp,
        o_mix, o_dnc, o_pred);

    // 2) weight conversions (independent of scan)
    k_f32_to_f16<<<dim3((2048 * 4096 + 255) / 256), dim3(256), 0, stream>>>(W1, W1h, 2048 * 4096);
    k_f32_to_f16<<<dim3((1024 * 2048 + 255) / 256), dim3(256), 0, stream>>>(W2, W2h, 1024 * 2048);

    // 3) x slice copy
    k_xslice<<<dim3((32 * 255 * 64 + 255) / 256), dim3(256), 0, stream>>>(x, o_xs);

    // 4) MLP layer 1: h1 = relu(GTA @ W1^T + b1), GTA = mix (f32 A staged in LDS)
    //    LDS: 2*32KB (A f32) + 2*16KB (B f16) = 96KB
    k_gemm_async<0><<<dim3(8192 / 128, 2048 / 128), dim3(256),
                      2 * (128 * 64 * 4) + 2 * (128 * 64 * 2), stream>>>(
        (const void*)o_mix, W1h, b1, h1, 8192, 2048, 4096);
    //    layer 2: h2 = relu(h1 @ W2^T + b2), A already f16 -> 64KB LDS
    k_gemm_async<1><<<dim3(8192 / 128, 1024 / 128), dim3(256),
                      2 * (128 * 64 * 2) + 2 * (128 * 64 * 2), stream>>>(
        (const void*)h1, W2h, b2, h2, 8192, 1024, 2048);

    // 5) logits = mean_t(h2 @ W3^T) + b3
    k_logits<<<dim3(32), dim3(256), 0, stream>>>(h2, W3, b3, o_logits);
}
